// MultiHeadAttention_61160334295194
// MI455X (gfx1250) — matrix operand order repaired and checked by reading, not compiled, not run on hardware
//
#include <hip/hip_runtime.h>
#include <hip/hip_bf16.h>
#include <math.h>

#define DM     1024
#define HEADS  16
#define DHEAD  64
#define BATCH  4
#define SEQ    2048
#define MTOT   (BATCH * SEQ)   // 8192

typedef __attribute__((ext_vector_type(16))) __bf16 bf16x16;
typedef __attribute__((ext_vector_type(8)))  float  f32x8;
typedef __attribute__((ext_vector_type(4)))  unsigned int u32x4;
typedef __attribute__((ext_vector_type(8)))  int          i32x8;
typedef __attribute__((ext_vector_type(4)))  int          i32x4;

union FragBF { bf16x16 v; uint4 q[2]; __bf16 h[16]; };

#if __has_builtin(__builtin_amdgcn_tensor_load_to_lds) && __has_builtin(__builtin_amdgcn_s_wait_tensorcnt)
#define HAVE_TDM 1
#else
#define HAVE_TDM 0
#endif

// ---------------------------------------------------------------------------
// WMMA fragment helpers (wave32 layouts per CDNA5 ISA 7.12.2)
// A 16x32 bf16: lane m = lane%16; K base = (lane<16 ? 0 : 8);
//               elems 0..7 -> K=base..base+7, elems 8..15 -> K=base+16..base+23
// B 32x16 bf16 staged as [n][k]: lane n = lane%16; K base = (lane<16 ? 0 : 8);
//               elems 0..7 -> K=base..base+7, elems 8..15 -> K=base+16..base+23
// C 16x16 f32: elem r -> row = r + (lane<16?0:8), col = lane%16
// ---------------------------------------------------------------------------
__device__ __forceinline__ bf16x16 ld_a_frag(const __bf16* t, int m0, int k0, int ld) {
  int lane = (int)(threadIdx.x & 31u);
  const __bf16* r = t + (size_t)(m0 + (lane & 15)) * ld + k0 + ((lane & 16) ? 8 : 0);
  FragBF f;
  f.q[0] = *(const uint4*)(r);
  f.q[1] = *(const uint4*)(r + 16);
  return f.v;
}

__device__ __forceinline__ bf16x16 ld_b_frag(const __bf16* t, int n0, int k0, int ld) {
  int lane = (int)(threadIdx.x & 31u);
  const __bf16* r = t + (size_t)(n0 + (lane & 15)) * ld + k0 + ((lane & 16) ? 8 : 0);
  FragBF f;
  f.q[0] = *(const uint4*)(r);
  f.q[1] = *(const uint4*)(r + 16);
  return f.v;
}

__device__ __forceinline__ f32x8 wmma_bf16(bf16x16 a, bf16x16 b, f32x8 c) {
  return __builtin_amdgcn_wmma_f32_16x16x32_bf16(
      /*neg_a=*/false, a, /*neg_b=*/false, b,
      /*c_mod=*/(short)0, c, /*reuse_a=*/false, /*reuse_b=*/false);
}

// ---------------------------------------------------------------------------
// DPP16 row_xmask reductions within a 16-lane row (wave32 half) — pure VALU.
// ---------------------------------------------------------------------------
template <int XM>
__device__ __forceinline__ float dpp_row_xmask(float x) {
  int v = __builtin_amdgcn_update_dpp(0, __float_as_int(x),
                                      0x160 + XM, 0xf, 0xf, true);
  return __int_as_float(v);
}
__device__ __forceinline__ float red_max16(float x) {
  x = fmaxf(x, dpp_row_xmask<1>(x));
  x = fmaxf(x, dpp_row_xmask<2>(x));
  x = fmaxf(x, dpp_row_xmask<4>(x));
  x = fmaxf(x, dpp_row_xmask<8>(x));
  return x;
}
__device__ __forceinline__ float red_sum16(float x) {
  x += dpp_row_xmask<1>(x);
  x += dpp_row_xmask<2>(x);
  x += dpp_row_xmask<4>(x);
  x += dpp_row_xmask<8>(x);
  return x;
}

#if HAVE_TDM
// ---------------------------------------------------------------------------
// Tensor Data Mover: 2-D tile load Global -> LDS with row padding.
// D# encoding per CDNA5 ISA (groups 0/1; groups 2/3
// zero for 2-D). data_size code 1 => 2-byte elements (bf16).
// pad_interval code c => pad after 2^(c+1) DWORDs; pad_amount code a => a+1
// DWORDs of padding added to the LDS address.
// This toolchain exposes the 6-arg builtin form:
//   (u32x4 g0, i32x8 g1, i32x4 g2, i32x4 g3, i32x8 extra, i32 cpol)
// ---------------------------------------------------------------------------
__device__ __forceinline__ void tdm_load_2d(const void* gaddr, unsigned lds_off,
                                            unsigned dim0, unsigned dim1,
                                            unsigned tile0, unsigned tile1,
                                            unsigned stride0,
                                            int pad_interval, int pad_amount) {
  unsigned long long ga = (unsigned long long)gaddr;
  u32x4 g0;
  g0[0] = 1u;                                  // count=1, no gather
  g0[1] = lds_off;                             // lds_addr (bytes)
  g0[2] = (unsigned)(ga & 0xffffffffull);      // global_addr[31:0]
  g0[3] = (unsigned)((ga >> 32) & 0x1ffffffull) | (2u << 30);  // [56:32], type=2

  i32x8 g1;
  unsigned w0 = (1u << 16);                    // data_size = 2 bytes
  if (pad_amount >= 0)
    w0 |= (1u << 20) | ((unsigned)pad_interval << 22) | ((unsigned)pad_amount << 25);
  g1[0] = (int)w0;
  g1[1] = (int)((dim0 & 0xffffu) << 16);                               // tensor_dim0[15:0]
  g1[2] = (int)(((dim0 >> 16) & 0xffffu) | ((dim1 & 0xffffu) << 16));  // dim0 hi | dim1 lo
  g1[3] = (int)(((dim1 >> 16) & 0xffffu) | ((tile0 & 0xffffu) << 16)); // dim1 hi | tile_dim0
  g1[4] = (int)(tile1 & 0xffffu);                                      // tile_dim1 (tile_dim2=0)
  g1[5] = (int)stride0;                                                // tensor_dim0_stride[31:0]
  g1[6] = 0;                                                           // stride0 hi | stride1 lo
  g1[7] = 0;
  i32x4 z4 = {0, 0, 0, 0};
  i32x8 z8 = {0, 0, 0, 0, 0, 0, 0, 0};
  __builtin_amdgcn_tensor_load_to_lds(g0, g1, z4, z4, z8, 0);
}
__device__ __forceinline__ unsigned lds_off_of(const void* p) {
  return (unsigned)(unsigned long long)p;   // generic->LDS aperture: low 32 bits
}
#endif

__device__ __forceinline__ void tdm_wait(int wave) {
#if HAVE_TDM
  if (wave == 0) __builtin_amdgcn_s_wait_tensorcnt(0);
#endif
}

// ---------------------------------------------------------------------------
// fp32 -> bf16 cast (vectorized x4)
// ---------------------------------------------------------------------------
__global__ void cast_f32_to_bf16_v4(const float* __restrict__ in,
                                    __bf16* __restrict__ out, int n4) {
  int i = blockIdx.x * blockDim.x + threadIdx.x;
  if (i >= n4) return;
  float4 x = ((const float4*)in)[i];
  union { __bf16 h[4]; uint2 u; } c;
  c.h[0] = (__bf16)x.x; c.h[1] = (__bf16)x.y;
  c.h[2] = (__bf16)x.z; c.h[3] = (__bf16)x.w;
  ((uint2*)out)[i] = c.u;
}

// ---------------------------------------------------------------------------
// GEMM: Y(M x 1024) = A(M x 1024, bf16 rowmajor) @ Bm(1024 x 1024, bf16 rowmajor)
// Block 128x128, BK=32, 256 threads = 8 waves arranged 4(M) x 2(N);
// each wave computes a 32x64 tile = 2x4 WMMA C frags.
// Double-buffered software pipeline: TDM (wave 0) stages A tile t+1 and all
// waves transpose-stage B tile t+1 while WMMAs consume tile t.
// MODE 0: store bf16 into head layout (B,H,S,64)   (for Q/K/V projections)
// MODE 1: store fp32 row-major (final output proj)
// ---------------------------------------------------------------------------
constexpr int G_BM = 128, G_BN = 128, G_BK = 32;
constexpr int G_LDA = G_BK + 8;   // 40: row = 64B + 16B pad (TDM pad 4 DW / 16 DW)
constexpr int G_LDB = G_BK + 8;

__device__ __forceinline__ void gemm_stage_A(const __bf16* A, int bm, int k0,
                                             __bf16* dst, int tid, int wave) {
#if HAVE_TDM
  (void)tid;
  if (wave == 0)
    tdm_load_2d(A + (size_t)bm * DM + k0, lds_off_of(dst),
                G_BK, G_BM, G_BK, G_BM, DM, /*pad_interval=*/3, /*pad_amount=*/3);
#else
  (void)wave;
  int idx = tid * 16;
  int row = idx >> 5;
  int col = idx & 31;
  const __bf16* g = A + (size_t)(bm + row) * DM + k0 + col;
  uint4 d0 = *(const uint4*)g;
  uint4 d1 = *(const uint4*)(g + 8);
  __bf16* s = dst + row * G_LDA + col;
  *(uint4*)s = d0;
  *(uint4*)(s + 8) = d1;
#endif
}

__device__ __forceinline__ void gemm_stage_BT(const __bf16* Bm, int k0, int bn,
                                              __bf16* dst, int tid) {
  int idx = tid * 16;
  int kr = idx >> 7;    // 0..31
  int nc = idx & 127;   // 0,16,...,112
  const __bf16* g = Bm + (size_t)(k0 + kr) * DM + bn + nc;
  FragBF f;
  f.q[0] = *(const uint4*)g;
  f.q[1] = *(const uint4*)(g + 8);
#pragma unroll
  for (int j = 0; j < 16; ++j) dst[(nc + j) * G_LDB + kr] = f.h[j];
}

template <int MODE>
__global__ __launch_bounds__(256)
void gemm_bf16_wmma(const __bf16* __restrict__ A,
                    const __bf16* __restrict__ Bm,
                    void* __restrict__ Out) {
  __shared__ __bf16 As[2 * G_BM * G_LDA];
  __shared__ __bf16 BsT[2 * G_BN * G_LDB];   // transposed tile [n][k]

  const int bm   = blockIdx.x * G_BM;
  const int bn   = blockIdx.y * G_BN;
  const int tid  = (int)threadIdx.x;
  const int wave = tid >> 5;
  const int wm   = wave & 3;   // 0..3 -> 32-row slab
  const int wn   = wave >> 2;  // 0..1 -> 64-col slab
  constexpr int NT = DM / G_BK;  // 32 k-tiles

  const f32x8 zero8 = {0.f, 0.f, 0.f, 0.f, 0.f, 0.f, 0.f, 0.f};
  f32x8 acc[2][4];
#pragma unroll
  for (int i = 0; i < 2; ++i)
#pragma unroll
    for (int j = 0; j < 4; ++j) acc[i][j] = zero8;

  // prologue: stage tile 0
  gemm_stage_A(A, bm, 0, As, tid, wave);
  gemm_stage_BT(Bm, 0, bn, BsT, tid);
  tdm_wait(wave);
  __syncthreads();

  for (int kt = 0; kt < NT; ++kt) {
    const int cur = kt & 1;
    const int nxt = cur ^ 1;
    if (kt + 1 < NT) {   // stage next tile while computing current
      gemm_stage_A(A, bm, (kt + 1) * G_BK, As + nxt * G_BM * G_LDA, tid, wave);
      gemm_stage_BT(Bm, (kt + 1) * G_BK, bn, BsT + nxt * G_BN * G_LDB, tid);
    }
    const __bf16* as = As + cur * G_BM * G_LDA;
    const __bf16* bs = BsT + cur * G_BN * G_LDB;
    bf16x16 afr0 = ld_a_frag(as, wm * 32 + 0,  0, G_LDA);
    bf16x16 afr1 = ld_a_frag(as, wm * 32 + 16, 0, G_LDA);
#pragma unroll
    for (int j = 0; j < 4; ++j) {
      bf16x16 bfr = ld_b_frag(bs, wn * 64 + j * 16, 0, G_LDB);
      acc[0][j] = wmma_bf16(afr0, bfr, acc[0][j]);
      acc[1][j] = wmma_bf16(afr1, bfr, acc[1][j]);
    }
    if (kt + 1 < NT) tdm_wait(wave);
    __syncthreads();
  }

  // epilogue
  const int lane = tid & 31;
  const int cn = lane & 15;
  const int rb = (lane & 16) ? 8 : 0;
#pragma unroll
  for (int i = 0; i < 2; ++i) {
#pragma unroll
    for (int j = 0; j < 4; ++j) {
#pragma unroll
      for (int r = 0; r < 8; ++r) {
        int grow = bm + wm * 32 + i * 16 + rb + r;
        int gcol = bn + wn * 64 + j * 16 + cn;
        float val = acc[i][j][r];
        if (MODE == 0) {
          int b = grow >> 11, s = grow & (SEQ - 1);
          int h = gcol >> 6, d = gcol & (DHEAD - 1);
          ((__bf16*)Out)[((size_t)(b * HEADS + h) * SEQ + s) * DHEAD + d] = (__bf16)val;
        } else {
          ((float*)Out)[(size_t)grow * DM + gcol] = val;
        }
      }
    }
  }
}

// ---------------------------------------------------------------------------
// Flash attention: grid (B*H, SEQ/128). 256 threads = 8 waves;
// each wave owns 16 query rows. Stream keys/values in 64-wide chunks,
// double-buffered: TDM stages K tile t+1 and all waves transpose-stage V
// tile t+1 while tile t is consumed by WMMAs + online softmax.
// Q/K/V in head layout (B,H,S,64) bf16; output written back to
// row-major (B*S, H*64) bf16 ready for the final projection GEMM.
// ---------------------------------------------------------------------------
constexpr int A_BC  = 64;
constexpr int A_LDT = DHEAD + 8;   // 72: row = 128B + 16B pad (TDM pad 4 DW / 32 DW)

__device__ __forceinline__ void attn_stage_K(const __bf16* Kb, int kc,
                                             __bf16* dst, int tid, int wave) {
#if HAVE_TDM
  (void)tid;
  if (wave == 0)
    tdm_load_2d(Kb + (size_t)kc * DHEAD, lds_off_of(dst),
                DHEAD, A_BC, DHEAD, A_BC, DHEAD, /*pad_interval=*/4, /*pad_amount=*/3);
#else
  (void)wave;
  int idx = tid * 16;
  int kr = idx >> 6;   // 0..63
  int dc = idx & 63;   // 0,16,32,48
  const __bf16* g = Kb + (size_t)(kc + kr) * DHEAD + dc;
  uint4 d0 = *(const uint4*)g;
  uint4 d1 = *(const uint4*)(g + 8);
  __bf16* s = dst + kr * A_LDT + dc;
  *(uint4*)s = d0;
  *(uint4*)(s + 8) = d1;
#endif
}

__device__ __forceinline__ void attn_stage_VT(const __bf16* Vb, int kc,
                                              __bf16* dst, int tid) {
  int idx = tid * 16;
  int vr = idx >> 6;
  int dc = idx & 63;
  const __bf16* g = Vb + (size_t)(kc + vr) * DHEAD + dc;
  FragBF f;
  f.q[0] = *(const uint4*)g;
  f.q[1] = *(const uint4*)(g + 8);
#pragma unroll
  for (int j = 0; j < 16; ++j) dst[(dc + j) * A_LDT + vr] = f.h[j];
}

__global__ __launch_bounds__(256)
void attn_flash_wmma(const __bf16* __restrict__ Qh, const __bf16* __restrict__ Kh,
                     const __bf16* __restrict__ Vh, __bf16* __restrict__ O) {
  __shared__ __bf16 Kt[2 * A_BC * A_LDT];       // [key][d]  (== B^T layout for QK^T)
  __shared__ __bf16 Vt[2 * DHEAD * A_LDT];      // [d][key]  (== B^T layout for P@V)
  __shared__ __bf16 Pbuf[8 * 16 * A_LDT];       // per-wave P staging (C->A relayout)

  const int bh   = (int)blockIdx.x;             // b*HEADS + h
  const int tid  = (int)threadIdx.x;
  const int wave = tid >> 5;
  const int lane = tid & 31;
  const int q0   = (int)blockIdx.y * 128 + wave * 16;
  constexpr int NT = SEQ / A_BC;                // 32 key chunks

  const __bf16* Qb = Qh + (size_t)bh * SEQ * DHEAD;
  const __bf16* Kb = Kh + (size_t)bh * SEQ * DHEAD;
  const __bf16* Vb = Vh + (size_t)bh * SEQ * DHEAD;

  // Q fragments live in registers for the whole kernel (rowmajor, ld = 64)
  bf16x16 aq0 = ld_a_frag(Qb, q0, 0,  DHEAD);
  bf16x16 aq1 = ld_a_frag(Qb, q0, 32, DHEAD);

  const f32x8 zero8 = {0.f, 0.f, 0.f, 0.f, 0.f, 0.f, 0.f, 0.f};
  f32x8 o[4];
#pragma unroll
  for (int j = 0; j < 4; ++j) o[j] = zero8;
  float m8[8], l8[8];
#pragma unroll
  for (int r = 0; r < 8; ++r) { m8[r] = -1e30f; l8[r] = 0.f; }

  const float scale = 0.125f;  // 1/sqrt(64)
  __bf16* Pw = Pbuf + wave * 16 * A_LDT;

  // prologue: stage chunk 0
  attn_stage_K(Kb, 0, Kt, tid, wave);
  attn_stage_VT(Vb, 0, Vt, tid);
  tdm_wait(wave);
  __syncthreads();

  for (int t = 0; t < NT; ++t) {
    const int cur = t & 1;
    const int nxt = cur ^ 1;
    if (t + 1 < NT) {   // stage next chunk while computing current
      attn_stage_K(Kb, (t + 1) * A_BC, Kt + nxt * A_BC * A_LDT, tid, wave);
      attn_stage_VT(Vb, (t + 1) * A_BC, Vt + nxt * DHEAD * A_LDT, tid);
    }
    const __bf16* ktc = Kt + cur * A_BC * A_LDT;
    const __bf16* vtc = Vt + cur * DHEAD * A_LDT;

    // scores: S = Q (16x64) @ K^T (64x64) -> 4 C frags over key tiles
    f32x8 sc[4];
#pragma unroll
    for (int j = 0; j < 4; ++j) {
      sc[j] = zero8;
      bf16x16 b0 = ld_b_frag(ktc, j * 16, 0,  A_LDT);
      bf16x16 b1 = ld_b_frag(ktc, j * 16, 32, A_LDT);
      sc[j] = wmma_bf16(aq0, b0, sc[j]);
      sc[j] = wmma_bf16(aq1, b1, sc[j]);
    }

    // online softmax (per C-frag element r == row within half-wave)
    float alpha[8];
#pragma unroll
    for (int r = 0; r < 8; ++r) {
      float mx = fmaxf(fmaxf(sc[0][r], sc[1][r]), fmaxf(sc[2][r], sc[3][r])) * scale;
      mx = red_max16(mx);
      float mn = fmaxf(m8[r], mx);
      alpha[r] = __expf(m8[r] - mn);
      m8[r] = mn;
      float ps = 0.f;
#pragma unroll
      for (int j = 0; j < 4; ++j) {
        float p = __expf(sc[j][r] * scale - mn);
        sc[j][r] = p;
        ps += p;
      }
      ps = red_sum16(ps);
      l8[r] = l8[r] * alpha[r] + ps;
#pragma unroll
      for (int j = 0; j < 4; ++j) o[j][r] *= alpha[r];
    }

    // P: C layout -> A layout via per-wave LDS round trip (in-order DS per wave)
    {
      int cn = lane & 15;
      int rb = (lane & 16) ? 8 : 0;
#pragma unroll
      for (int j = 0; j < 4; ++j)
#pragma unroll
        for (int r = 0; r < 8; ++r)
          Pw[(rb + r) * A_LDT + j * 16 + cn] = (__bf16)sc[j][r];
    }
    bf16x16 ap0 = ld_a_frag(Pw, 0, 0,  A_LDT);
    bf16x16 ap1 = ld_a_frag(Pw, 0, 32, A_LDT);
#pragma unroll
    for (int j = 0; j < 4; ++j) {
      bf16x16 bv0 = ld_b_frag(vtc, j * 16, 0,  A_LDT);
      bf16x16 bv1 = ld_b_frag(vtc, j * 16, 32, A_LDT);
      o[j] = wmma_bf16(ap0, bv0, o[j]);
      o[j] = wmma_bf16(ap1, bv1, o[j]);
    }

    if (t + 1 < NT) tdm_wait(wave);
    __syncthreads();
  }

  // epilogue: O /= l, write row-major (B*S, H*64) bf16
  {
    int cn = lane & 15;
    int rb = (lane & 16) ? 8 : 0;
    int b = bh >> 4;
    int h = bh & (HEADS - 1);
#pragma unroll
    for (int j = 0; j < 4; ++j) {
#pragma unroll
      for (int r = 0; r < 8; ++r) {
        int row = q0 + rb + r;       // s
        int col = j * 16 + cn;       // d
        float inv = 1.0f / l8[r];
        O[(size_t)(b * SEQ + row) * DM + h * DHEAD + col] = (__bf16)(o[j][r] * inv);
      }
    }
  }
}

// ---------------------------------------------------------------------------
// Host orchestration
// ---------------------------------------------------------------------------
extern "C" void kernel_launch(void* const* d_in, const int* in_sizes, int n_in,
                              void* d_out, int out_size, void* d_ws, size_t ws_size,
                              hipStream_t stream) {
  (void)in_sizes; (void)n_in; (void)out_size; (void)ws_size;
  const float* q  = (const float*)d_in[0];
  const float* kx = (const float*)d_in[1];
  const float* vx = (const float*)d_in[2];
  const float* Wq = (const float*)d_in[3];
  const float* Wk = (const float*)d_in[4];
  const float* Wv = (const float*)d_in[5];
  const float* Wo = (const float*)d_in[6];

  char* ws = (char*)d_ws;
  size_t off = 0;
  auto take = [&](size_t elems) -> __bf16* {
    __bf16* p = (__bf16*)(ws + off);
    off += ((elems * sizeof(__bf16)) + 255) & ~(size_t)255;
    return p;
  };
  const size_t ACT = (size_t)MTOT * DM;   // 8M elems
  const size_t WEL = (size_t)DM * DM;     // 1M elems
  __bf16* qb  = take(ACT);
  __bf16* kb  = take(ACT);
  __bf16* vb  = take(ACT);
  __bf16* Wqb = take(WEL);
  __bf16* Wkb = take(WEL);
  __bf16* Wvb = take(WEL);
  __bf16* Wob = take(WEL);
  __bf16* Qh  = take(ACT);  // (B,H,S,64)
  __bf16* Kh  = take(ACT);
  __bf16* Vh  = take(ACT);
  __bf16* Ob  = take(ACT);  // (B*S, H*64) row-major

  auto castN = [&](const float* src, __bf16* dst, size_t n) {
    int n4 = (int)(n / 4);
    cast_f32_to_bf16_v4<<<(n4 + 255) / 256, 256, 0, stream>>>(src, dst, n4);
  };
  castN(q,  qb,  ACT);
  castN(kx, kb,  ACT);
  castN(vx, vb,  ACT);
  castN(Wq, Wqb, WEL);
  castN(Wk, Wkb, WEL);
  castN(Wv, Wvb, WEL);
  castN(Wo, Wob, WEL);

  dim3 gp(MTOT / 128, DM / 128);  // 64 x 8
  gemm_bf16_wmma<0><<<gp, 256, 0, stream>>>(qb, Wqb, (void*)Qh);
  gemm_bf16_wmma<0><<<gp, 256, 0, stream>>>(kb, Wkb, (void*)Kh);
  gemm_bf16_wmma<0><<<gp, 256, 0, stream>>>(vb, Wvb, (void*)Vh);

  attn_flash_wmma<<<dim3(BATCH * HEADS, SEQ / 128), 256, 0, stream>>>(Qh, Kh, Vh, Ob);

  gemm_bf16_wmma<1><<<gp, 256, 0, stream>>>(Ob, Wob, d_out);
}
